// Quantize_60103772340315
// MI455X (gfx1250) — matrix operand order repaired and checked by reading, not compiled, not run on hardware
//
#include <hip/hip_runtime.h>
#include <hip/hip_bf16.h>

// VQ codebook quantization for MI455X (gfx1250, wave32, WMMA f16 16x16x32).
// Pipeline: prep (codebook -> (-2x)-scaled f16 A-fragments + e_sq + zero hist)
//        -> main (WMMA computes esq - 2*X.E^T directly via C-accumulator init,
//                 register argmin, gather+write, partial MSE)
//        -> final (deterministic diff reduction + perplexity).

typedef __attribute__((ext_vector_type(16))) _Float16 v16h;
typedef __attribute__((ext_vector_type(8)))  float    v8f;

#define Q_ELEMS  16777216   // 32*128*64*64 quantized floats
#define DIFF_OFF 16777216   // scalar diff
#define IDX_OFF  16777217   // 32*2*64*64 indices (as float)
#define PP_OFF   17039361   // 2 perplexities

// ---- workspace layout (bytes) ----
// [0, 131072)        : f16 codebook * (-2), pre-swizzled into WMMA A-fragments
// [131072, 135168)   : e_sq, 1024 floats
// [135168, 139264)   : histogram counts, 1024 ints
// [139264, 147456)   : per-block diff partials, 2048 floats

__device__ __forceinline__ v16h ld_frag_lds(const _Float16* p) {
  union { uint4 q[2]; v16h v; } u;
  u.q[0] = *(const uint4*)(p);
  u.q[1] = *(const uint4*)(p + 8);
  return u.v;
}
__device__ __forceinline__ v16h ld_frag_g(const uint4* p) {
  union { uint4 q[2]; v16h v; } u;
  u.q[0] = p[0];
  u.q[1] = p[1];
  return u.v;
}

// ---------------- prep: one thread per (n, code) ----------------
__global__ __launch_bounds__(256) void vq_prep(const float* __restrict__ embed,
                                               _Float16* __restrict__ efrag,
                                               float* __restrict__ esq,
                                               int* __restrict__ counts) {
  int gid = blockIdx.x * 256 + threadIdx.x;      // grid = 4 blocks -> 1024 threads
  int n = gid >> 9, c = gid & 511;
  const float* er = embed + (size_t)gid * 64;
  int kt = c >> 4, lanelow = c & 15;
  float s = 0.f;
  for (int d = 0; d < 64; ++d) {
    float v = er[d];
    s += v * v;
    // A 16x32 layout (contiguous, matches B): lane-half h holds K in {h*16..h*16+15}
    int dd   = d & 31;
    int h    = dd >> 4;
    int j    = dd & 15;
    int dblk = d >> 5;
    int lane = lanelow + h * 16;
    // store -2*e so the WMMA accumulator directly produces esq - 2*<x,e>
    efrag[n * 32768 + (((kt * 2 + dblk) * 32 + lane) << 4) + j] = (_Float16)(-2.0f * v);
  }
  esq[gid] = s;
  counts[gid] = 0;   // re-init histogram every call (ws not re-poisoned)
}

// ---------------- main: one block = 128 pixels of one codebook ----------------
__global__ __launch_bounds__(256) void vq_main(const float* __restrict__ x,
                                               const float* __restrict__ embed,
                                               const _Float16* __restrict__ efrag,
                                               const float* __restrict__ esq_g,
                                               int* __restrict__ counts,
                                               float* __restrict__ partials,
                                               float* __restrict__ out) {
  __shared__ _Float16 x_lds[128 * 72];   // padded stride: 144B rows, 16B aligned
  __shared__ float    esq[512];
  __shared__ int      idx_lds[128];
  __shared__ float    red[256];

  const int t  = threadIdx.x;
  const int n  = blockIdx.y;
  const int m0 = blockIdx.x * 128;       // tiles never cross a batch (4096 % 128 == 0)
  const int b  = m0 >> 12;
  const int hw0 = m0 & 4095;
  const long xbase = (long)(b * 128 + n * 64) * 4096 + hw0;

  for (int i = t; i < 512; i += 256) esq[i] = esq_g[n * 512 + i];
  // coalesced float4 load of X^T tile (4 pixels, same d), f32 -> f16, transpose into LDS
  for (int i = t; i < 2048; i += 256) {
    int d = i >> 5, p4 = (i & 31) << 2;
    float4 v = *(const float4*)&x[xbase + (long)d * 4096 + p4];
    x_lds[(p4 + 0) * 72 + d] = (_Float16)v.x;
    x_lds[(p4 + 1) * 72 + d] = (_Float16)v.y;
    x_lds[(p4 + 2) * 72 + d] = (_Float16)v.z;
    x_lds[(p4 + 3) * 72 + d] = (_Float16)v.w;
  }
  __syncthreads();

  const int wave = t >> 5, lane = t & 31;
  const int half = lane >> 4, col = lane & 15;
  const int p0 = wave * 16;
  const int p  = p0 + col;

  // B fragments (X^T, 32d x 16px): lane-half holds K range {0..15}/{16..31}
  v16h bf0 = ld_frag_lds(&x_lds[p * 72 + half * 16]);        // d 0..31
  v16h bf1 = ld_frag_lds(&x_lds[p * 72 + 32 + half * 16]);   // d 32..63

  const uint4* fa = (const uint4*)(efrag + n * 32768);

  float bestv = 3.4e38f;
  int   bestk = 0;
  for (int kt = 0; kt < 32; ++kt) {
    v16h a0 = ld_frag_g(fa + (((kt * 2 + 0) * 32 + lane) << 1));
    v16h a1 = ld_frag_g(fa + (((kt * 2 + 1) * 32 + lane) << 1));
    // C accumulator pre-loaded with esq: VGPR r <-> code row (kt*16 + half*8 + r)
    const int kb = kt * 16 + half * 8;
    float4 e0 = *(const float4*)&esq[kb];
    float4 e1 = *(const float4*)&esq[kb + 4];
    v8f acc;
    acc[0] = e0.x; acc[1] = e0.y; acc[2] = e0.z; acc[3] = e0.w;
    acc[4] = e1.x; acc[5] = e1.y; acc[6] = e1.z; acc[7] = e1.w;
    acc = __builtin_amdgcn_wmma_f32_16x16x32_f16(false, a0, false, bf0, (short)0, acc, false, false);
    acc = __builtin_amdgcn_wmma_f32_16x16x32_f16(false, a1, false, bf1, (short)0, acc, false, false);
    // acc[r] == esq[kb+r] - 2*<x_p, e_{kb+r}>  (score), lane col = pixel
#pragma unroll
    for (int r = 0; r < 8; ++r) {
      float s = acc[r];
      if (s < bestv) { bestv = s; bestk = kb + r; }   // ascending k => first-index ties
    }
  }
  // merge the two code halves (lane L <-> L^16 share the same pixel)
  {
    float ov = __shfl_xor(bestv, 16, 32);
    int   ok = __shfl_xor(bestk, 16, 32);
    if (ov < bestv || (ov == bestv && ok < bestk)) { bestv = ov; bestk = ok; }
  }
  if (lane < 16) idx_lds[p0 + lane] = bestk;
  __syncthreads();

  // indices (as float, reference layout [B, N, H, W]) + histogram
  if (t < 128) {
    int k = idx_lds[t];
    out[IDX_OFF + (long)(b * 2 + n) * 4096 + (hw0 + t)] = (float)k;
    atomicAdd(&counts[n * 512 + k], 1);
  }

  // quantized gather (f32 codebook, L2-hot) + partial MSE; float4 stores, coalesced per-d
  float dacc = 0.f;
  for (int i = t; i < 2048; i += 256) {
    int d = i >> 5, p4 = (i & 31) << 2;
    long gi = xbase + (long)d * 4096 + p4;
    float4 xv = *(const float4*)&x[gi];
    float4 q;
    q.x = embed[((n * 512 + idx_lds[p4 + 0]) << 6) + d];
    q.y = embed[((n * 512 + idx_lds[p4 + 1]) << 6) + d];
    q.z = embed[((n * 512 + idx_lds[p4 + 2]) << 6) + d];
    q.w = embed[((n * 512 + idx_lds[p4 + 3]) << 6) + d];
    float d0 = q.x - xv.x, d1 = q.y - xv.y, d2 = q.z - xv.z, d3 = q.w - xv.w;
    dacc += d0 * d0 + d1 * d1 + d2 * d2 + d3 * d3;
    *(float4*)&out[gi] = q;
  }
  red[t] = dacc;
  __syncthreads();
  for (int s = 128; s > 0; s >>= 1) { if (t < s) red[t] += red[t + s]; __syncthreads(); }
  if (t == 0) partials[blockIdx.y * 1024 + blockIdx.x] = red[0];
}

// ---------------- final: deterministic diff + perplexity ----------------
__global__ __launch_bounds__(256) void vq_final(const int* __restrict__ counts,
                                                const float* __restrict__ partials,
                                                float* __restrict__ out) {
  __shared__ float red[256];
  int t = threadIdx.x;
  float local = 0.f;
  for (int i = t; i < 2048; i += 256) local += partials[i];
  red[t] = local;
  __syncthreads();
  for (int s = 128; s > 0; s >>= 1) { if (t < s) red[t] += red[t + s]; __syncthreads(); }
  if (t == 0) out[DIFF_OFF] = red[0] * (1.0f / 16777216.0f);
  for (int n = 0; n < 2; ++n) {
    __syncthreads();
    float l = 0.f;
    for (int k = t; k < 512; k += 256) {
      float pp = (float)counts[n * 512 + k] * (1.0f / 131072.0f);
      l += pp * logf(pp + 1e-10f);
    }
    red[t] = l;
    __syncthreads();
    for (int s = 128; s > 0; s >>= 1) { if (t < s) red[t] += red[t + s]; __syncthreads(); }
    if (t == 0) out[PP_OFF + n] = expf(-red[0]);
  }
}

extern "C" void kernel_launch(void* const* d_in, const int* in_sizes, int n_in,
                              void* d_out, int out_size, void* d_ws, size_t ws_size,
                              hipStream_t stream) {
  const float* x     = (const float*)d_in[0];   // [32, 128, 64, 64] f32
  const float* embed = (const float*)d_in[1];   // [2, 512, 64] f32
  float* out = (float*)d_out;

  char* ws = (char*)d_ws;
  _Float16* efrag    = (_Float16*)ws;              // 131072 B
  float*    esq      = (float*)(ws + 131072);      // 4096 B
  int*      counts   = (int*)(ws + 135168);        // 4096 B
  float*    partials = (float*)(ws + 139264);      // 8192 B

  vq_prep<<<dim3(4), dim3(256), 0, stream>>>(embed, efrag, esq, counts);
  vq_main<<<dim3(1024, 2), dim3(256), 0, stream>>>(x, embed, efrag, esq, counts, partials, out);
  vq_final<<<dim3(1), dim3(256), 0, stream>>>(counts, partials, out);
}
